// GrammarModel_47304769798332
// MI455X (gfx1250) — matrix operand order repaired and checked by reading, not compiled, not run on hardware
//
#include <hip/hip_runtime.h>
#include <hip/hip_bf16.h>

#define BATCH 64
#define LEN   128
#define NT    64     // terminals (K dim)
#define NN    128    // nonterminals
#define BG    16     // batches per workgroup (WMMA N tile)

typedef __attribute__((ext_vector_type(16))) __bf16 v16bf;
typedef __attribute__((ext_vector_type(8)))  __bf16 v8bf;
typedef __attribute__((ext_vector_type(8)))  float  v8f;

__device__ __forceinline__ float sigmoidf(float v) {
    return 1.0f / (1.0f + __expf(-v));
}

__device__ __forceinline__ float clip01(float v) {
    return __builtin_fminf(__builtin_fmaxf(v, 0.0f), 1.0f);
}

// Build a 16x32 bf16 A-fragment for one lane.
// Contiguous K order (matches B): lanes 0-15 hold row M=lane, K = 0..15;
// lanes 16-31 hold row M=lane-16, K = 16..31.
// p points at ntp_row + tblock + (laneHi ? 16 : 0); contiguous bf16 over t.
__device__ __forceinline__ v16bf load_a_frag(const __bf16* p) {
    v8bf lo = *(const v8bf*)(p);        // K offsets +0..+7
    v8bf hi = *(const v8bf*)(p + 8);    // K offsets +8..+15
    v16bf r;
#pragma unroll
    for (int j = 0; j < 8; ++j) { r[j] = lo[j]; r[j + 8] = hi[j]; }
    return r;
}

// Precompute: sntp = bf16(sigmoid(nonterminal_productions)), 128*128*64 elems.
__global__ void sigmoid_to_bf16(const float* __restrict__ in,
                                __bf16* __restrict__ out, int n) {
    int i = blockIdx.x * blockDim.x + threadIdx.x;
    if (i < n) out[i] = (__bf16)sigmoidf(i < n ? in[i] : 0.0f);
}

// One workgroup = 16 batches, 1024 threads = 32 waves, full 127-step scan.
__global__ __launch_bounds__(1024)
void grammar_scan(const float* __restrict__ x,      // [B][LEN][NT]
                  const float* __restrict__ termp,  // [NN][NT]
                  const __bf16* __restrict__ sntp,  // [NN*NN][NT] sigmoid, bf16
                  float* __restrict__ out)          // [B]
{
    __shared__ float  prevbuf[2][BG][NN + 1];   // +1 pad: kill 64-bank conflicts
    __shared__ __bf16 xT[BG][NT + 8];           // x_i transposed to [b][t], padded
    __shared__ float  xmask[BG];
    __shared__ float  outs[BG];

    const int b0     = blockIdx.x * BG;
    const int tid    = threadIdx.x;
    const int wave   = tid >> 5;
    const int lane   = tid & 31;
    const int lane15 = lane & 15;
    const int laneHi = lane >> 4;          // which half-wave
    const int kb     = laneHi * 16;        // per-lane K sub-block for A frags

    // ---- init: tpc[b,n] = sum_t sigmoid(termp[n,t]) * x[b,0,t]  (not clipped)
    for (int idx = tid; idx < BG * NN; idx += 1024) {
        int bl = idx >> 7;
        int n  = idx & (NN - 1);
        const float* xr = x + (size_t)(b0 + bl) * LEN * NT;   // time step 0
        const float* tr = termp + n * NT;
        float s = 0.0f;
#pragma unroll 4
        for (int t = 0; t < NT; ++t) s = fmaf(sigmoidf(tr[t]), xr[t], s);
        prevbuf[0][bl][n] = s;
    }
    __syncthreads();
    if (tid < BG) outs[tid] = prevbuf[0][tid][0];
    __syncthreads();

    // ---- sequential scan over time steps 1..127
    for (int i = 1; i < LEN; ++i) {
        const int curb = i & 1;
        const int prvb = curb ^ 1;

        // stage x_i into LDS as bf16, transposed to [b][t]; 1024 threads == BG*NT
        {
            int bl = tid >> 6;
            int t  = tid & (NT - 1);
            float v = x[(size_t)(b0 + bl) * LEN * NT + (size_t)i * NT + t];
            xT[bl][t] = (__bf16)v;
            if (t == NT - 1) xmask[bl] = v;
        }
        __syncthreads();

        // Opaque zero offset (uniform -> SGPR), refreshed every step: defeats
        // cross-step CSE/hoist of the step-invariant A-fragment loads (which
        // previously spilled 2KB/lane to scratch) WITHOUT destroying the
        // global address-space provenance of sntp, so the loads stay
        // global_load_b128 (LOADcnt only, saddr form) instead of flat_load.
        size_t zoff = 0;
        asm volatile("" : "+s"(zoff));
        const __bf16* sntp_i = sntp + zoff;

        // B fragments (32x16, K x batch): depend only on lane, not on n/m-tile.
        // lanes 0-15: K = 0..15 (+tblock); lanes 16-31: K = 16..31 (+tblock).
        const __bf16* xrow = &xT[lane15][0];
        v16bf bf0 = *(const v16bf*)(xrow + laneHi * 16);        // K block 0..31
        v16bf bf1 = *(const v16bf*)(xrow + 32 + laneHi * 16);   // K block 32..63

        const float* pr = &prevbuf[prvb][lane15][0];

#pragma unroll
        for (int j = 0; j < 4; ++j) {
            const int n = wave * 4 + j;
            const __bf16* abase = sntp_i + (size_t)n * NN * NT;

            float acc = 0.0f;
#pragma unroll
            for (int mt = 0; mt < 8; ++mt) {
                const int m0 = mt * 16;
                const __bf16* arow = abase + (size_t)(m0 + lane15) * NT + kb;
                __builtin_prefetch(arow + 16 * NT, 0, 3);
                v16bf a0 = load_a_frag(arow);        // t = 0..31
                v16bf a1 = load_a_frag(arow + 32);   // t = 32..63
                v8f c = {};
                c = __builtin_amdgcn_wmma_f32_16x16x32_bf16(
                        false, a0, false, bf0, (short)0, c, false, false);
                c = __builtin_amdgcn_wmma_f32_16x16x32_bf16(
                        false, a1, false, bf1, (short)0, c, false, false);
                // D layout: lane col = batch b0+lane15, VGPR v = row m0 + laneHi*8 + v
                const int rb = m0 + laneHi * 8;
#pragma unroll
                for (int v = 0; v < 8; ++v)
                    acc = fmaf(clip01(c[v]), pr[rb + v], acc);
            }
            // combine the two half-wave row groups; deterministic order
            float tot = acc + __shfl_xor(acc, 16, 32);
            if (lane < 16) prevbuf[curb][lane][n] = clip01(tot);
        }
        __syncthreads();

        if (tid < BG) {
            float m = xmask[tid];
            outs[tid] = outs[tid] * m + prevbuf[curb][tid][0] * (1.0f - m);
        }
        __syncthreads();
    }

    if (tid < BG) out[b0 + tid] = outs[tid];
}

extern "C" void kernel_launch(void* const* d_in, const int* in_sizes, int n_in,
                              void* d_out, int out_size, void* d_ws, size_t ws_size,
                              hipStream_t stream) {
    (void)in_sizes; (void)n_in; (void)out_size; (void)ws_size;
    const float* x    = (const float*)d_in[0];   // (64,128,64)  f32
    const float* tp   = (const float*)d_in[1];   // (128,64)     f32
    const float* ntp  = (const float*)d_in[2];   // (128,128,64) f32
    __bf16*      sntp = (__bf16*)d_ws;           // 1,048,576 bf16 = 2 MB scratch

    const int nelem = NN * NN * NT;
    sigmoid_to_bf16<<<(nelem + 255) / 256, 256, 0, stream>>>(ntp, sntp, nelem);
    grammar_scan<<<BATCH / BG, 1024, 0, stream>>>(x, tp, sntp, (float*)d_out);
}
